// DynamicFilter_82214263980650
// MI455X (gfx1250) — matrix operand order repaired and checked by reading, not compiled, not run on hardware
//
#include <hip/hip_runtime.h>

typedef __attribute__((ext_vector_type(16))) _Float16 v16h;
typedef __attribute__((ext_vector_type(8)))  float    v8f;

#define TILE 16
#define HREG 20              // TILE + 2*2 halo for conv2
#define IREG 24              // HREG + 2*2 halo for conv1
#define HPIX (HREG * HREG)   // 400
#define IPIX (IREG * IREG)   // 576
// h buffer pitch: 34 halves = 17 dwords per pixel (coprime with 64 banks)

// K position held by `j`-th half of a lane's A operand (contiguous K order):
// lane<16: K 0..15 ; lane>=16: K 16..31
__device__ __forceinline__ int a_klocal(int hi, int j) {
    return hi * 16 + j;
}

// low 32 bits of a generic pointer to __shared__ = wave-relative LDS byte offset
__device__ __forceinline__ unsigned lds_off_u(const void* p) {
    return (unsigned)(unsigned long long)p;
}

__launch_bounds__(256)
__global__ void dynfilt_fused(const float* __restrict__ image,
                              const float* __restrict__ refer,
                              const float* __restrict__ w1,
                              const float* __restrict__ b1,
                              const float* __restrict__ w2,
                              const float* __restrict__ b2,
                              float* __restrict__ out)
{
    __shared__ __align__(16) unsigned xin_u[IPIX + 1];    // f16 pairs + zero sentinel
    __shared__ __align__(16) float    ximg[IPIX];         // f32 image for stage 3
    __shared__ __align__(16) unsigned hbuf_u[HPIX * 17];  // h f16 [pix][ch] pitch 17d (27200B)
    __shared__ __align__(32) _Float16 w1A[2 * 2 * 32 * 16]; // conv1 A, pre-swizzled
    __shared__ __align__(32) _Float16 w2A[25 * 32 * 16];    // conv2 A, pre-swizzled
    // hbuf region doubles as an f32 staging buffer for the async input loads
    float* staging = (float*)hbuf_u;                      // [0..575]=img, [576..1151]=ref

    const int tid  = threadIdx.x;
    const int lane = tid & 31;
    const int wave = tid >> 5;
    const int hi   = (lane >> 4) & 1;
    const int lm   = lane & 15;

    const int gx0 = blockIdx.x * TILE;
    const int gy0 = blockIdx.y * TILE;
    const int b   = blockIdx.z;
    const float* imgB = image + (size_t)b * 512 * 512;
    const float* refB = refer + (size_t)b * 512 * 512;

    // -------- async load input tile (halo 4) into LDS staging; OOB -> 0 -----
    for (int pix = tid; pix < IPIX; pix += 256) {
        int iy = pix / IREG, ix = pix % IREG;
        int gy = gy0 - 4 + iy, gx = gx0 - 4 + ix;
        if ((unsigned)gy < 512u && (unsigned)gx < 512u) {
            unsigned l0 = lds_off_u(&staging[pix]);
            unsigned l1 = lds_off_u(&staging[IPIX + pix]);
            unsigned long long ga0 = (unsigned long long)(imgB + gy * 512 + gx);
            unsigned long long ga1 = (unsigned long long)(refB + gy * 512 + gx);
            asm volatile("global_load_async_to_lds_b32 %0, %1, off"
                         :: "v"(l0), "v"(ga0) : "memory");
            asm volatile("global_load_async_to_lds_b32 %0, %1, off"
                         :: "v"(l1), "v"(ga1) : "memory");
        } else {
            staging[pix] = 0.f;
            staging[IPIX + pix] = 0.f;
        }
    }

    // ------------- pre-swizzle conv1 weights into A layout [s][mt][lane][16]
    // GEMM1 K-index = tap*2 + ch  (K = 50, padded to 64)
    for (int idx = tid; idx < 2 * 2 * 32 * 16; idx += 256) {
        int j  = idx & 15;
        int ln = (idx >> 4) & 31;
        int mt = (idx >> 9) & 1;
        int s  = (idx >> 10) & 1;
        int k  = s * 32 + a_klocal((ln >> 4) & 1, j);
        int m  = mt * 16 + (ln & 15);
        _Float16 val = (_Float16)0.f;
        if (k < 50) {
            int tap = k >> 1, ch = k & 1;
            val = (_Float16)w1[(m * 2 + ch) * 25 + tap];
        }
        w1A[idx] = val;
    }

    // ------------- pre-swizzle conv2 weights into A layout [tap][lane][16]
    for (int idx = tid; idx < 25 * 32 * 16; idx += 256) {
        int j  = idx & 15;
        int ln = (idx >> 4) & 31;
        int t  = idx >> 9;
        int ch = a_klocal((ln >> 4) & 1, j);
        int m  = ln & 15;
        _Float16 val = (_Float16)0.f;
        if (m < 9)
            val = (_Float16)w2[(m * 32 + ch) * 25 + t];
        w2A[idx] = val;
    }

    // wait for this wave's async-to-LDS transfers, then convert f32 -> f16
    asm volatile("s_wait_asynccnt 0x0" ::: "memory");
    for (int pix = tid; pix < IPIX; pix += 256) {
        float vi = staging[pix];
        float vr = staging[IPIX + pix];
        union { _Float16 h[2]; unsigned u; } pk;
        pk.h[0] = (_Float16)vi; pk.h[1] = (_Float16)vr;
        xin_u[pix] = pk.u;
        ximg[pix]  = vi;
    }
    if (tid == 0) xin_u[IPIX] = 0u;   // zero sentinel for padded K reads

    __syncthreads();

    // ===== stage 1: h = LeakyReLU(conv1(x)+b1) on the 20x20 halo region =====
    // One B operand per 16-pixel tile feeds both 16-channel M-tiles.
    // B K-pair offsets (dwords in xin_u) for taps tb..tb+7, tb = s*16+hi*8:
    //   tb=0 : {0,1,2,3,4,24,25,26}      tb=8 : {27,28,48,49,50,51,52,72}
    //   tb=16: {73,74,75,76,96,97,98,99} tb=24: {100, sentinel x7}
    static const int offT0[8]  = {0, 1, 2, 3, 4, 24, 25, 26};
    static const int offT8[8]  = {27, 28, 48, 49, 50, 51, 52, 72};
    static const int offT16[8] = {73, 74, 75, 76, 96, 97, 98, 99};

    for (int nt = wave; nt < 25; nt += 8) {
        int p  = nt * 16 + lm;                 // halo pixel 0..399
        int hy = p / HREG, hx = p % HREG;
        int ib = hy * IREG + hx;               // dword base in xin_u

        v8f c0, c1;
        #pragma unroll
        for (int r = 0; r < 8; ++r) {
            c0[r] = b1[hi * 8 + r];
            c1[r] = b1[16 + hi * 8 + r];
        }

        #pragma unroll
        for (int s = 0; s < 2; ++s) {
            union { v16h h; unsigned u[8]; } B;
            #pragma unroll
            for (int v = 0; v < 8; ++v) {
                unsigned idx;
                if (s == 0) {
                    idx = ib + (hi ? offT8[v] : offT0[v]);
                } else {
                    idx = (v == 0) ? (unsigned)(ib + (hi ? 100 : offT16[0]))
                                   : (hi ? (unsigned)IPIX : (unsigned)(ib + offT16[v]));
                }
                B.u[v] = xin_u[idx];
            }
            const v16h a0 = *(const v16h*)&w1A[((s * 2 + 0) * 32 + lane) * 16];
            const v16h a1 = *(const v16h*)&w1A[((s * 2 + 1) * 32 + lane) * 16];
            c0 = __builtin_amdgcn_wmma_f32_16x16x32_f16(false, a0, false, B.h,
                                                        (short)0, c0, false, false);
            c1 = __builtin_amdgcn_wmma_f32_16x16x32_f16(false, a1, false, B.h,
                                                        (short)0, c1, false, false);
        }

        // LeakyReLU(0.1), cvt f16; zero h outside the image (conv2 zero-pad)
        int gy = gy0 - 2 + hy, gx = gx0 - 2 + hx;
        bool inb = ((unsigned)gy < 512u) && ((unsigned)gx < 512u);
        #pragma unroll
        for (int mt = 0; mt < 2; ++mt) {
            int chd = mt * 8 + hi * 4;         // dword offset of channel base
            #pragma unroll
            for (int r = 0; r < 8; r += 2) {
                float v0 = mt ? c1[r]     : c0[r];
                float v1 = mt ? c1[r + 1] : c0[r + 1];
                v0 = (v0 >= 0.f) ? v0 : 0.1f * v0;
                v1 = (v1 >= 0.f) ? v1 : 0.1f * v1;
                union { _Float16 h[2]; unsigned u; } pk;
                pk.h[0] = inb ? (_Float16)v0 : (_Float16)0.f;
                pk.h[1] = inb ? (_Float16)v1 : (_Float16)0.f;
                hbuf_u[p * 17 + chd + (r >> 1)] = pk.u;
            }
        }
    }

    __syncthreads();

    // ===== stage 2: filters = conv2(h)+b2 ; stage 3: dynamic 3x3 apply =====
    // wave w owns output rows 2w and 2w+1; both rows share every A tile.
    {
        const int oy0 = wave * 2, ox = lm;
        const int base0 = (oy0 * HREG + ox) * 17 + hi * 8;
        const int base1 = base0 + HREG * 17;

        v8f c0, c1;
        #pragma unroll
        for (int r = 0; r < 8; ++r) {
            int m = hi * 8 + r;
            float bv = (m < 9) ? b2[m] : 0.f;
            c0[r] = bv; c1[r] = bv;
        }

        #pragma unroll
        for (int t = 0; t < 25; ++t) {         // K = 800 = 25 taps x 32 ch
            const int doff = ((t / 5) * HREG + (t % 5)) * 17;  // compile-time
            union { v16h h; unsigned u[8]; } B0, B1;
            #pragma unroll
            for (int v = 0; v < 8; ++v) B0.u[v] = hbuf_u[base0 + doff + v];
            #pragma unroll
            for (int v = 0; v < 8; ++v) B1.u[v] = hbuf_u[base1 + doff + v];
            const v16h a = *(const v16h*)&w2A[(t * 32 + lane) * 16];
            c0 = __builtin_amdgcn_wmma_f32_16x16x32_f16(false, a, false, B0.h,
                                                        (short)0, c0, false, false);
            c1 = __builtin_amdgcn_wmma_f32_16x16x32_f16(false, a, false, B1.h,
                                                        (short)0, c1, false, false);
        }

        // C layout: lane n holds m=0..7 of pixel n; lane n+16 holds m=8..15.
        float f8a = __shfl(c0[0], lm + 16, 32);
        float f8b = __shfl(c1[0], lm + 16, 32);
        if (hi == 0) {
            float acc0 = 0.f, acc1 = 0.f;
            #pragma unroll
            for (int m = 0; m < 8; ++m) {
                int di = m / 3 - 1, dj = m % 3 - 1;
                acc0 += c0[m] * ximg[(oy0 + 4 + di) * IREG + (ox + 4 + dj)];
                acc1 += c1[m] * ximg[(oy0 + 5 + di) * IREG + (ox + 4 + dj)];
            }
            acc0 += f8a * ximg[(oy0 + 5) * IREG + (ox + 5)];  // m=8 -> (+1,+1)
            acc1 += f8b * ximg[(oy0 + 6) * IREG + (ox + 5)];
            size_t o = ((size_t)b * 512 + (gy0 + oy0)) * 512 + (gx0 + ox);
            out[o]       = acc0;
            out[o + 512] = acc1;
        }
    }
}

extern "C" void kernel_launch(void* const* d_in, const int* in_sizes, int n_in,
                              void* d_out, int out_size, void* d_ws, size_t ws_size,
                              hipStream_t stream) {
    const float* image = (const float*)d_in[0];
    const float* refer = (const float*)d_in[1];
    const float* w1    = (const float*)d_in[2];
    const float* b1    = (const float*)d_in[3];
    const float* w2    = (const float*)d_in[4];
    const float* b2    = (const float*)d_in[5];
    float* out = (float*)d_out;

    dim3 grid(512 / TILE, 512 / TILE, 8);   // 32 x 32 tiles x 8 batch
    dynfilt_fused<<<grid, 256, 0, stream>>>(image, refer, w1, b1, w2, b2, out);
}
